// Top_Down_34677565948532
// MI455X (gfx1250) — matrix operand order repaired and checked by reading, not compiled, not run on hardware
//
#include <hip/hip_runtime.h>
#include <hip/hip_bf16.h>
#include <stdint.h>

typedef __attribute__((ext_vector_type(16))) _Float16 v16h;
typedef __attribute__((ext_vector_type(8)))  float    v8f;

#define BATCH 8
#define CH    256
#define KCH   128
#define NCLS  21
#define NPIX  16384
#define EPSGN 1e-5f

// ----------------------------------------------------------------------------
// WMMA fragment loaders (gfx1250, wave32, 16x16x32 f16)
// A 16x32: lane m=L&15, half h=L>>4; VGPR0..3 -> K=8h+{0..7}; VGPR4..7 -> K=16+8h+{0..7}
// B 32x16: lane n=L&15, half h=L>>4; VGPR0..3 -> K=8h+{0..7}; VGPR4..7 -> K=16+8h+{0..7}
// ----------------------------------------------------------------------------
__device__ __forceinline__ v16h load_fragA(const _Float16* __restrict__ row, int kbase, int h) {
  union { v16h v; uint32_t u[8]; } f;
  const uint32_t* p = (const uint32_t*)row;
  int b0 = (kbase >> 1) + h * 4;      // half offset kbase + 8h
  int b1 = b0 + 8;                    // half offset kbase + 8h + 16
#pragma unroll
  for (int j = 0; j < 4; ++j) { f.u[j] = p[b0 + j]; f.u[4 + j] = p[b1 + j]; }
  return f.v;
}

__device__ __forceinline__ v16h load_fragB(const _Float16* __restrict__ row, int kbase, int h) {
  union { v16h v; uint32_t u[8]; } f;
  const uint32_t* p = (const uint32_t*)row;
  int b0 = (kbase >> 1) + h * 4;      // half offset kbase + 8h
  int b1 = b0 + 8;                    // half offset kbase + 8h + 16
#pragma unroll
  for (int j = 0; j < 4; ++j) { f.u[j] = p[b0 + j]; f.u[4 + j] = p[b1 + j]; }
  return f.v;
}

// ----------------------------------------------------------------------------
// Generic batched WMMA GEMM:  D[b][m][n] = scale * sum_k A[b][m][k] * B[b][n][k]
// (optionally a 2nd operand pair appended in K: used for the concat conv)
// Each wave computes a 16(M) x 64(N) strip = 4 tiles: A fragment reused 4x and
// 4 independent accumulator chains give the XDL back-to-back WMMA ILP.
// Output stored "N-major, M-contiguous" f16 (channel-last for convs).
// Optional GroupNorm stats: per-batch atomic (sum, sumsq).
// 8 waves / block; (numM*numN64) % 8 == 0 so a block never spans batches.
// ----------------------------------------------------------------------------
__global__ __launch_bounds__(256) void k_gemm(
    const _Float16* __restrict__ A1, int lda1, int K1,
    const _Float16* __restrict__ A2, int lda2, int K2,
    long long sA,
    const _Float16* __restrict__ B1, int ldb1, long long sB1,
    const _Float16* __restrict__ B2, int ldb2, long long sB2,
    _Float16* __restrict__ Cout, int ldc, long long sC,
    int numM, int numN64, float scale, float* __restrict__ stats)
{
  __shared__ float rs[256];
  __shared__ float rq[256];
  int tid  = threadIdx.x;
  int wave = (blockIdx.x * 256 + tid) >> 5;
  int lane = tid & 31;
  int tpb  = numM * numN64;
  int b    = wave / tpb;
  int t    = wave - b * tpb;
  int mt   = t / numN64;
  int nt   = t - mt * numN64;
  int l16  = lane & 15;
  int h    = lane >> 4;
  int col0 = nt * 64 + l16;

  const _Float16* ar = A1 + (long long)b * sA + (long long)(mt * 16 + l16) * lda1;
  const _Float16* br0 = B1 + (long long)b * sB1 + (long long)(col0     ) * ldb1;
  const _Float16* br1 = B1 + (long long)b * sB1 + (long long)(col0 + 16) * ldb1;
  const _Float16* br2 = B1 + (long long)b * sB1 + (long long)(col0 + 32) * ldb1;
  const _Float16* br3 = B1 + (long long)b * sB1 + (long long)(col0 + 48) * ldb1;

  v8f acc0 = {}, acc1 = {}, acc2 = {}, acc3 = {};
  for (int k = 0; k < K1; k += 32) {
    v16h af = load_fragA(ar, k, h);
    v16h bf0 = load_fragB(br0, k, h);
    v16h bf1 = load_fragB(br1, k, h);
    v16h bf2 = load_fragB(br2, k, h);
    v16h bf3 = load_fragB(br3, k, h);
    acc0 = __builtin_amdgcn_wmma_f32_16x16x32_f16(false, af, false, bf0, (short)0, acc0, false, false);
    acc1 = __builtin_amdgcn_wmma_f32_16x16x32_f16(false, af, false, bf1, (short)0, acc1, false, false);
    acc2 = __builtin_amdgcn_wmma_f32_16x16x32_f16(false, af, false, bf2, (short)0, acc2, false, false);
    acc3 = __builtin_amdgcn_wmma_f32_16x16x32_f16(false, af, false, bf3, (short)0, acc3, false, false);
  }
  if (K2 > 0) {
    const _Float16* ar2 = A2 + (long long)b * sA + (long long)(mt * 16 + l16) * lda2;
    const _Float16* cr0 = B2 + (long long)b * sB2 + (long long)(col0     ) * ldb2;
    const _Float16* cr1 = B2 + (long long)b * sB2 + (long long)(col0 + 16) * ldb2;
    const _Float16* cr2 = B2 + (long long)b * sB2 + (long long)(col0 + 32) * ldb2;
    const _Float16* cr3 = B2 + (long long)b * sB2 + (long long)(col0 + 48) * ldb2;
    for (int k = 0; k < K2; k += 32) {
      v16h af = load_fragA(ar2, k, h);
      v16h bf0 = load_fragB(cr0, k, h);
      v16h bf1 = load_fragB(cr1, k, h);
      v16h bf2 = load_fragB(cr2, k, h);
      v16h bf3 = load_fragB(cr3, k, h);
      acc0 = __builtin_amdgcn_wmma_f32_16x16x32_f16(false, af, false, bf0, (short)0, acc0, false, false);
      acc1 = __builtin_amdgcn_wmma_f32_16x16x32_f16(false, af, false, bf1, (short)0, acc1, false, false);
      acc2 = __builtin_amdgcn_wmma_f32_16x16x32_f16(false, af, false, bf2, (short)0, acc2, false, false);
      acc3 = __builtin_amdgcn_wmma_f32_16x16x32_f16(false, af, false, bf3, (short)0, acc3, false, false);
    }
  }

  // D layout: lane holds column n, rows m = 8h + r (r = 0..7, consecutive)
  float s = 0.f, q = 0.f;
  long long cbase = (long long)b * sC + mt * 16 + h * 8;
  v8f accs[4] = {acc0, acc1, acc2, acc3};
#pragma unroll
  for (int j = 0; j < 4; ++j) {
    union { _Float16 hh[8]; uint32_t u[4]; } ov;
#pragma unroll
    for (int r = 0; r < 8; ++r) {
      float v = accs[j][r] * scale;
      ov.hh[r] = (_Float16)v;
      s += v; q += v * v;
    }
    uint32_t* cp = (uint32_t*)(Cout + cbase + (long long)(col0 + j * 16) * ldc);
#pragma unroll
    for (int w = 0; w < 4; ++w) cp[w] = ov.u[w];
  }

  if (stats) {  // all waves in a block share b (tpb % 8 == 0)
    rs[tid] = s; rq[tid] = q;
    __syncthreads();
    for (int st = 128; st > 0; st >>= 1) {
      if (tid < st) { rs[tid] += rs[tid + st]; rq[tid] += rq[tid + st]; }
      __syncthreads();
    }
    if (tid == 0) {
      atomicAdd(&stats[2 * b],     rs[0]);
      atomicAdd(&stats[2 * b + 1], rq[0]);
    }
  }
}

// ----------------------------------------------------------------------------
// Small helper kernels
// ----------------------------------------------------------------------------
__global__ void k_zero(float* p, int n) {
  int i = blockIdx.x * blockDim.x + threadIdx.x;
  if (i < n) p[i] = 0.f;
}

__global__ void k_f2h(const float* __restrict__ s, _Float16* __restrict__ d, int n) {
  int i = blockIdx.x * blockDim.x + threadIdx.x;
  if (i < n) d[i] = (_Float16)s[i];
}

// feats [b][C][n] f32 -> featsT [b][n][C] f16   (32x32 LDS tiles)
__global__ __launch_bounds__(256) void k_transpose_cn(const float* __restrict__ src,
                                                      _Float16* __restrict__ dst)
{
  __shared__ float tile[32][33];
  int blk = blockIdx.x;
  int tpb = (CH / 32) * (NPIX / 32);
  int b   = blk / tpb;
  int r   = blk % tpb;
  int c0  = (r / (NPIX / 32)) * 32;
  int n0  = (r % (NPIX / 32)) * 32;
  int tid = threadIdx.x;
#pragma unroll
  for (int e = 0; e < 4; ++e) {
    int idx = tid + e * 256;
    int ci = idx >> 5, nj = idx & 31;
    tile[ci][nj] = src[((long long)(b * CH + c0 + ci)) * NPIX + n0 + nj];
  }
  __syncthreads();
#pragma unroll
  for (int e = 0; e < 4; ++e) {
    int idx = tid + e * 256;
    int ni = idx >> 5, cj = idx & 31;
    dst[((long long)(b * NPIX + n0 + ni)) * CH + c0 + cj] = (_Float16)tile[cj][ni];
  }
}

// bilinear upsample 64x64 -> 128x128, align_corners=True
__global__ void k_interp(const float* __restrict__ src, float* __restrict__ dst) {
  int idx = blockIdx.x * blockDim.x + threadIdx.x;   // exact grid: 8*21*16384
  int xw = idx & 127;
  int yh = (idx >> 7) & 127;
  int ck = idx >> 14;
  const float* s = src + (long long)ck * 4096;
  const float sc = 63.0f / 127.0f;
  float fy = yh * sc, fx = xw * sc;
  int y0 = (int)fy, x0 = (int)fx;
  int y1 = min(y0 + 1, 63), x1 = min(x0 + 1, 63);
  float wy = fy - y0, wx = fx - x0;
  float r0 = s[y0 * 64 + x0] * (1.f - wx) + s[y0 * 64 + x1] * wx;
  float r1 = s[y1 * 64 + x0] * (1.f - wx) + s[y1 * 64 + x1] * wx;
  dst[idx] = r0 * (1.f - wy) + r1 * wy;
}

__device__ __forceinline__ float blk_sum256(float v, float* red) {
  int tid = threadIdx.x;
  red[tid] = v;
  __syncthreads();
  for (int st = 128; st > 0; st >>= 1) {
    if (tid < st) red[tid] += red[tid + st];
    __syncthreads();
  }
  float r = red[0];
  __syncthreads();
  return r;
}

// per-row softmax over 16384 spatial positions; one block per (b,class) row
__global__ __launch_bounds__(256) void k_softmax_rows(float* __restrict__ p) {
  __shared__ float red[256];
  long long base = (long long)blockIdx.x * NPIX;
  int tid = threadIdx.x;
  float m = -1e30f;
  for (int i = tid; i < NPIX; i += 256) m = fmaxf(m, p[base + i]);
  red[tid] = m; __syncthreads();
  for (int st = 128; st > 0; st >>= 1) {
    if (tid < st) red[tid] = fmaxf(red[tid], red[tid + st]);
    __syncthreads();
  }
  m = red[0]; __syncthreads();
  float s = 0.f;
  for (int i = tid; i < NPIX; i += 256) s += expf(p[base + i] - m);
  s = blk_sum256(s, red);
  float inv = 1.0f / s;
  for (int i = tid; i < NPIX; i += 256) p[base + i] = expf(p[base + i] - m) * inv;
}

// ctx[b][c][k] = sum_n probs[b][k][n] * feats[b][c][n]   (one wave per output)
__global__ __launch_bounds__(256) void k_ctx(const float* __restrict__ feats,
                                             const float* __restrict__ probs,
                                             float* __restrict__ ctx)
{
  int gw   = (blockIdx.x * 256 + threadIdx.x) >> 5;
  int lane = threadIdx.x & 31;
  int b = gw / (CH * NCLS);
  int r = gw % (CH * NCLS);
  int c = r / NCLS, k = r % NCLS;
  const float* fr = feats + ((long long)(b * CH + c)) * NPIX;
  const float* pr = probs + ((long long)(b * NCLS + k)) * NPIX;
  float acc = 0.f;
  for (int i = lane; i < NPIX; i += 32) acc = fmaf(fr[i], pr[i], acc);
  for (int off = 16; off > 0; off >>= 1) acc += __shfl_xor(acc, off, 32);
  if (lane == 0) ctx[((long long)(b * CH + c)) * 32 + k] = acc;
}

__device__ void gn_relu_shared(float* arr, int L, const float* __restrict__ g,
                               const float* __restrict__ bt, float* red)
{
  int tid = threadIdx.x;
  float s = 0.f, q = 0.f;
  for (int i = tid; i < L; i += 256) { float x = arr[i]; s += x; q += x * x; }
  float S = blk_sum256(s, red);
  float Q = blk_sum256(q, red);
  float mean = S / (float)L;
  float rstd = rsqrtf(Q / (float)L - mean * mean + EPSGN);
  for (int i = tid; i < L; i += 256) {
    int o = i / NCLS;
    arr[i] = fmaxf(0.f, (arr[i] - mean) * rstd * g[o] + bt[o]);
  }
  __syncthreads();
}

// k/v branch over the 21 context "pixels" — one block per batch, all fp32 in LDS
__global__ __launch_bounds__(256) void k_kv(
    const float* __restrict__ ctx,
    const float* __restrict__ fo_w1, const float* __restrict__ fo_g1, const float* __restrict__ fo_b1,
    const float* __restrict__ fo_w2, const float* __restrict__ fo_g2, const float* __restrict__ fo_b2,
    const float* __restrict__ fd_w,  const float* __restrict__ fd_g,  const float* __restrict__ fd_b,
    _Float16* __restrict__ kh, _Float16* __restrict__ vh)
{
  __shared__ float ctx_s[CH * NCLS];
  __shared__ float sA[KCH * NCLS];
  __shared__ float sB[KCH * NCLS];
  __shared__ float red[256];
  int b = blockIdx.x, tid = threadIdx.x;

  for (int i = tid; i < CH * NCLS; i += 256) {
    int c = i / NCLS, k = i % NCLS;
    ctx_s[i] = ctx[((long long)(b * CH + c)) * 32 + k];
  }
  __syncthreads();

  // t1 = fo_w1 @ ctx
  for (int i = tid; i < KCH * NCLS; i += 256) {
    int o = i / NCLS, k = i % NCLS;
    float a = 0.f;
    for (int c = 0; c < CH; ++c) a = fmaf(fo_w1[o * CH + c], ctx_s[c * NCLS + k], a);
    sA[i] = a;
  }
  __syncthreads();
  gn_relu_shared(sA, KCH * NCLS, fo_g1, fo_b1, red);

  // t2 = fo_w2 @ t1  -> kmat
  for (int i = tid; i < KCH * NCLS; i += 256) {
    int o = i / NCLS, k = i % NCLS;
    float a = 0.f;
    for (int c = 0; c < KCH; ++c) a = fmaf(fo_w2[o * KCH + c], sA[c * NCLS + k], a);
    sB[i] = a;
  }
  __syncthreads();
  gn_relu_shared(sB, KCH * NCLS, fo_g2, fo_b2, red);

  // kh[b][kk][c] (K-major rows, c contiguous), zero-pad kk=21..31
  for (int i = tid; i < 32 * KCH; i += 256) {
    int kk = i / KCH, c = i % KCH;
    kh[(long long)b * 32 * KCH + i] = (kk < NCLS) ? (_Float16)sB[c * NCLS + kk] : (_Float16)0.f;
  }

  // v = fd_w @ ctx
  for (int i = tid; i < KCH * NCLS; i += 256) {
    int o = i / NCLS, k = i % NCLS;
    float a = 0.f;
    for (int c = 0; c < CH; ++c) a = fmaf(fd_w[o * CH + c], ctx_s[c * NCLS + k], a);
    sA[i] = a;
  }
  __syncthreads();
  gn_relu_shared(sA, KCH * NCLS, fd_g, fd_b, red);

  // vh[b][c][kk] (c rows, k contiguous), zero-pad kk=21..31
  for (int i = tid; i < KCH * 32; i += 256) {
    int c = i / 32, kk = i % 32;
    vh[(long long)b * KCH * 32 + i] = (kk < NCLS) ? (_Float16)sA[c * NCLS + kk] : (_Float16)0.f;
  }
}

// softmax over the 21 valid attention classes; one thread per (b,pixel) row of 32
__global__ void k_softmax21(_Float16* __restrict__ sim) {
  int idx = blockIdx.x * blockDim.x + threadIdx.x;
  if (idx >= BATCH * NPIX) return;
  _Float16* row = sim + (long long)idx * 32;
  float v[NCLS];
  float m = -1e30f;
#pragma unroll
  for (int k = 0; k < NCLS; ++k) { v[k] = (float)row[k]; m = fmaxf(m, v[k]); }
  float s = 0.f;
#pragma unroll
  for (int k = 0; k < NCLS; ++k) { v[k] = expf(v[k] - m); s += v[k]; }
  float inv = 1.0f / s;
#pragma unroll
  for (int k = 0; k < NCLS; ++k) row[k] = (_Float16)(v[k] * inv);
#pragma unroll
  for (int k = NCLS; k < 32; ++k) row[k] = (_Float16)0.f;
}

// GroupNorm(1 group) + ReLU, channel-last f16 -> f16
__global__ void k_gn(const _Float16* __restrict__ pre, _Float16* __restrict__ out,
                     const float* __restrict__ stats, const float* __restrict__ g,
                     const float* __restrict__ bt, int C, float invCount, long long total)
{
  long long idx = (long long)blockIdx.x * blockDim.x + threadIdx.x;
  if (idx >= total) return;
  long long perB = (long long)C * NPIX;
  int b = (int)(idx / perB);
  int c = (int)(idx % C);
  float mean = stats[2 * b] * invCount;
  float rstd = rsqrtf(stats[2 * b + 1] * invCount - mean * mean + EPSGN);
  float x = (float)pre[idx];
  out[idx] = (_Float16)fmaxf(0.f, (x - mean) * rstd * g[c] + bt[c]);
}

// final GN+ReLU with transpose back to [b][C][H][W] f32 (32x32 LDS tiles)
__global__ __launch_bounds__(256) void k_final(const _Float16* __restrict__ pre,
                                               const float* __restrict__ stats,
                                               const float* __restrict__ g,
                                               const float* __restrict__ bt,
                                               float* __restrict__ out)
{
  __shared__ float tile[32][33];
  int blk = blockIdx.x;
  int tpb = (CH / 32) * (NPIX / 32);
  int b   = blk / tpb;
  int r   = blk % tpb;
  int c0  = (r / (NPIX / 32)) * 32;
  int n0  = (r % (NPIX / 32)) * 32;
  int tid = threadIdx.x;
  const float invCount = 1.0f / ((float)CH * (float)NPIX);
  float mean = stats[2 * b] * invCount;
  float rstd = rsqrtf(stats[2 * b + 1] * invCount - mean * mean + EPSGN);
#pragma unroll
  for (int e = 0; e < 4; ++e) {
    int idx = tid + e * 256;
    int ni = idx >> 5, cj = idx & 31;
    float x = (float)pre[((long long)(b * NPIX + n0 + ni)) * CH + c0 + cj];
    tile[cj][ni] = fmaxf(0.f, (x - mean) * rstd * g[c0 + cj] + bt[c0 + cj]);
  }
  __syncthreads();
#pragma unroll
  for (int e = 0; e < 4; ++e) {
    int idx = tid + e * 256;
    int ci = idx >> 5, nj = idx & 31;
    out[((long long)(b * CH + c0 + ci)) * NPIX + n0 + nj] = tile[ci][nj];
  }
}

// ----------------------------------------------------------------------------
extern "C" void kernel_launch(void* const* d_in, const int* in_sizes, int n_in,
                              void* d_out, int out_size, void* d_ws, size_t ws_size,
                              hipStream_t stream)
{
  (void)in_sizes; (void)n_in; (void)out_size; (void)ws_size;
  const float* feats  = (const float*)d_in[0];
  const float* outaux = (const float*)d_in[1];
  const float* fp_w1  = (const float*)d_in[2];
  const float* fp_g1  = (const float*)d_in[3];
  const float* fp_b1  = (const float*)d_in[4];
  const float* fp_w2  = (const float*)d_in[5];
  const float* fp_g2  = (const float*)d_in[6];
  const float* fp_b2  = (const float*)d_in[7];
  const float* fo_w1  = (const float*)d_in[8];
  const float* fo_g1  = (const float*)d_in[9];
  const float* fo_b1  = (const float*)d_in[10];
  const float* fo_w2  = (const float*)d_in[11];
  const float* fo_g2  = (const float*)d_in[12];
  const float* fo_b2  = (const float*)d_in[13];
  const float* fd_w   = (const float*)d_in[14];
  const float* fd_g   = (const float*)d_in[15];
  const float* fd_b   = (const float*)d_in[16];
  const float* fu_w   = (const float*)d_in[17];
  const float* fu_g   = (const float*)d_in[18];
  const float* fu_b   = (const float*)d_in[19];
  const float* cb_w   = (const float*)d_in[20];
  const float* cb_g   = (const float*)d_in[21];
  const float* cb_b   = (const float*)d_in[22];
  float* out = (float*)d_out;

  char* wsb = (char*)d_ws;
  size_t off = 0;
  auto alloc = [&](size_t bytes) -> char* {
    char* p = wsb + off;
    off = (off + bytes + 255) & ~(size_t)255;
    return p;
  };

  float*    probs  = (float*)   alloc((size_t)BATCH * NCLS * NPIX * 4);   // 11 MB
  _Float16* featsT = (_Float16*)alloc((size_t)BATCH * NPIX * CH * 2);     // 67 MB
  _Float16* wh1    = (_Float16*)alloc((size_t)KCH * CH * 2);
  _Float16* wh2    = (_Float16*)alloc((size_t)KCH * KCH * 2);
  _Float16* wfu    = (_Float16*)alloc((size_t)CH * KCH * 2);
  _Float16* wcb    = (_Float16*)alloc((size_t)CH * 2 * CH * 2);
  float*    ctx    = (float*)   alloc((size_t)BATCH * CH * 32 * 4);
  _Float16* kh     = (_Float16*)alloc((size_t)BATCH * 32 * KCH * 2);
  _Float16* vh     = (_Float16*)alloc((size_t)BATCH * KCH * 32 * 2);
  _Float16* preA   = (_Float16*)alloc((size_t)BATCH * NPIX * KCH * 2);    // 33.5 MB
  _Float16* preB   = (_Float16*)alloc((size_t)BATCH * NPIX * KCH * 2);    // 33.5 MB
  _Float16* simh   = (_Float16*)alloc((size_t)BATCH * NPIX * 32 * 2);     // 8.4 MB
  _Float16* upA    = (_Float16*)alloc((size_t)BATCH * NPIX * CH * 2);     // 67 MB
  _Float16* upB    = (_Float16*)alloc((size_t)BATCH * NPIX * CH * 2);     // 67 MB
  float*    stats  = (float*)   alloc(64 * 4);

  const long long sPix128 = (long long)NPIX * KCH;
  const long long sPix256 = (long long)NPIX * CH;
  const long long sPix32  = (long long)NPIX * 32;
  const int N64 = NPIX / 64;   // 256 N-supertiles per batch

  // stage setup
  k_zero<<<1, 64, 0, stream>>>(stats, 64);
  k_f2h<<<(KCH * CH + 255) / 256, 256, 0, stream>>>(fp_w1, wh1, KCH * CH);
  k_f2h<<<(KCH * KCH + 255) / 256, 256, 0, stream>>>(fp_w2, wh2, KCH * KCH);
  k_f2h<<<(CH * KCH + 255) / 256, 256, 0, stream>>>(fu_w, wfu, CH * KCH);
  k_f2h<<<(CH * 2 * CH + 255) / 256, 256, 0, stream>>>(cb_w, wcb, CH * 2 * CH);
  k_transpose_cn<<<BATCH * (CH / 32) * (NPIX / 32), 256, 0, stream>>>(feats, featsT);

  // SpatialGather
  k_interp<<<(BATCH * NCLS * NPIX) / 256, 256, 0, stream>>>(outaux, probs);
  k_softmax_rows<<<BATCH * NCLS, 256, 0, stream>>>(probs);
  k_ctx<<<(BATCH * CH * NCLS) / 8, 256, 0, stream>>>(feats, probs, ctx);

  // k / v branch (tiny)
  k_kv<<<BATCH, 256, 0, stream>>>(ctx, fo_w1, fo_g1, fo_b1, fo_w2, fo_g2, fo_b2,
                                  fd_w, fd_g, fd_b, kh, vh);

  // q1 = fp_w1 @ feats  (M=128, N=16384, K=256)
  k_gemm<<<BATCH * 8 * N64 / 8, 256, 0, stream>>>(
      wh1, CH, CH, nullptr, 0, 0, 0LL,
      featsT, CH, sPix256, nullptr, 0, 0LL,
      preA, KCH, sPix128, 8, N64, 1.f, stats + 0);
  k_gn<<<(unsigned)(BATCH * sPix128 / 256), 256, 0, stream>>>(
      preA, preB, stats + 0, fp_g1, fp_b1, KCH, 1.f / ((float)KCH * NPIX), BATCH * sPix128);

  // q2 = fp_w2 @ q1  (K=128)
  k_gemm<<<BATCH * 8 * N64 / 8, 256, 0, stream>>>(
      wh2, KCH, KCH, nullptr, 0, 0, 0LL,
      preB, KCH, sPix128, nullptr, 0, 0LL,
      preA, KCH, sPix128, 8, N64, 1.f, stats + 16);
  k_gn<<<(unsigned)(BATCH * sPix128 / 256), 256, 0, stream>>>(
      preA, preB, stats + 16, fp_g2, fp_b2, KCH, 1.f / ((float)KCH * NPIX), BATCH * sPix128);

  // sim[n][kk] = (q . k) * kc^-0.5  (M=32 classes, N=16384, K=128)
  k_gemm<<<BATCH * 2 * N64 / 8, 256, 0, stream>>>(
      kh, KCH, KCH, nullptr, 0, 0, (long long)32 * KCH,
      preB, KCH, sPix128, nullptr, 0, 0LL,
      simh, 32, sPix32, 2, N64, 0.08838834764831845f, nullptr);
  k_softmax21<<<(BATCH * NPIX) / 256, 256, 0, stream>>>(simh);

  // att = v @ sim^T  (M=128, N=16384, K=32 — one WMMA per accumulator)
  k_gemm<<<BATCH * 8 * N64 / 8, 256, 0, stream>>>(
      vh, 32, 32, nullptr, 0, 0, (long long)KCH * 32,
      simh, 32, sPix32, nullptr, 0, 0LL,
      preA, KCH, sPix128, 8, N64, 1.f, nullptr);

  // up = fu_w @ att  (M=256, N=16384, K=128)
  k_gemm<<<BATCH * 16 * N64 / 8, 256, 0, stream>>>(
      wfu, KCH, KCH, nullptr, 0, 0, 0LL,
      preA, KCH, sPix128, nullptr, 0, 0LL,
      upA, CH, sPix256, 16, N64, 1.f, stats + 32);
  k_gn<<<(unsigned)(BATCH * sPix256 / 256), 256, 0, stream>>>(
      upA, upB, stats + 32, fu_g, fu_b, CH, 1.f / ((float)CH * NPIX), BATCH * sPix256);

  // fused conv over concat: cb = cb_w[:, :256] @ up + cb_w[:, 256:] @ feats
  k_gemm<<<BATCH * 16 * N64 / 8, 256, 0, stream>>>(
      wcb, 2 * CH, CH, wcb + CH, 2 * CH, CH, 0LL,
      upB, CH, sPix256, featsT, CH, sPix256,
      upA, CH, sPix256, 16, N64, 1.f, stats + 48);

  // final GN + ReLU, transpose back to [b][C][H][W] f32
  k_final<<<BATCH * (CH / 32) * (NPIX / 32), 256, 0, stream>>>(upA, stats + 48, cb_g, cb_b, out);
}
